// thesis_nca_model_1357209665884
// MI455X (gfx1250) — matrix operand order repaired and checked by reading, not compiled, not run on hardware
//
#include <hip/hip_runtime.h>
#include <hip/hip_bf16.h>

// MI455X / gfx1250: wave32, WMMA 16x16x32 f16 (f32 accum), TDM async tensor loads.
// Compute-bound NCA update: perception (depthwise 3^3 conv) -> 64->128 relu -> 128->16,
// fused with alive max-pool mask, stochastic residual update, angle mod 2pi.
//
// Roofline: ~21 GFLOP vs ~132 MB minimum traffic (x is L2-resident: 64 MB < 192 MB L2)
// -> heavily compute-bound on the pointwise matmuls -> they run on v_wmma f16/f32.
// Halo staging for the perception conv uses the Tensor Data Mover (double-buffered,
// pad_enable gives the 66-dword LDS row stride for free) when available.

typedef __attribute__((ext_vector_type(16))) _Float16 v16h;
typedef __attribute__((ext_vector_type(8)))  float    v8f;
typedef unsigned int u32x4_t __attribute__((ext_vector_type(4)));
typedef int          i32x4_t __attribute__((ext_vector_type(4)));
typedef int          i32x8_t __attribute__((ext_vector_type(8)));

#ifndef __has_builtin
#define __has_builtin(x) 0
#endif
#if __has_builtin(__builtin_amdgcn_tensor_load_to_lds) && \
    __has_builtin(__builtin_amdgcn_s_wait_tensorcnt)
#define HAS_TDM 1
#else
#define HAS_TDM 0
#endif

#define C_CH 16
#define HID  128
#define DD   64
#define DD2  (64*64)
#define DD3  (64*64*64)
#define TWO_PI_F 6.28318530717958647692f

union FragH {
  v16h v;
  _Float16 h[16];
  uint4 q[2];
};

// A-fragment (16x32 f16) from LDS, contiguous K order (matches B fragment):
//   lanes 0-15: K = kbase + [0..15]
//   lanes 16-31: same rows, K = kbase + 16 + [0..15].
__device__ __forceinline__ v16h load_a_frag_lds(const _Float16* base, int row0,
                                                int stride, int kbase, int lane) {
  FragH f;
  const int r = lane & 15, g = lane >> 4;
  const _Float16* p = base + (row0 + r) * stride + kbase + g * 16;
  f.q[0] = *(const uint4*)(p);
  f.q[1] = *(const uint4*)(p + 8);
  return f.v;
}

// B-fragment (32x16 f16) built from an f32 weight matrix W[row][K]:
//   lane n (=lane&15) holds column n; halves split K at g*16.
//   B[k][n] = W[rowbase+n][kbase + g*16 + i] for i in [0,16).
__device__ __forceinline__ v16h load_b_frag_global(const float* __restrict__ w,
                                                   int rowbase, int K, int kbase,
                                                   int lane) {
  FragH f;
  const int n = lane & 15, g = lane >> 4;
  const float* p = w + (size_t)(rowbase + n) * K + kbase + g * 16;
  const float4 f0 = ((const float4*)p)[0];
  const float4 f1 = ((const float4*)p)[1];
  const float4 f2 = ((const float4*)p)[2];
  const float4 f3 = ((const float4*)p)[3];
  f.h[0]  = (_Float16)f0.x; f.h[1]  = (_Float16)f0.y;
  f.h[2]  = (_Float16)f0.z; f.h[3]  = (_Float16)f0.w;
  f.h[4]  = (_Float16)f1.x; f.h[5]  = (_Float16)f1.y;
  f.h[6]  = (_Float16)f1.z; f.h[7]  = (_Float16)f1.w;
  f.h[8]  = (_Float16)f2.x; f.h[9]  = (_Float16)f2.y;
  f.h[10] = (_Float16)f2.z; f.h[11] = (_Float16)f2.w;
  f.h[12] = (_Float16)f3.x; f.h[13] = (_Float16)f3.y;
  f.h[14] = (_Float16)f3.z; f.h[15] = (_Float16)f3.w;
  return f.v;
}

#if HAS_TDM
// Issue one TDM load of a 64(w) x 4(h) x 3(d) f32 tile into LDS at ldsByteAddr.
// pad_enable: after every 64 dwords (pad_interval=5) skip 2 dwords (pad_amount=1)
// -> LDS row stride 66 dwords; skipped pad slots keep their pre-zeroed halo value.
// D# layout per CDNA5 ISA ch. 7/8 (group0: flags/lds/global/type, group1: dims).
__device__ __forceinline__ void tdm_load_halo(const float* gptr, unsigned ldsByteAddr) {
  const unsigned long long ga = (unsigned long long)(size_t)gptr;
  u32x4_t g0;
  g0[0] = 1u;                                        // count=1, no gather
  g0[1] = ldsByteAddr;                               // lds_addr (bytes)
  g0[2] = (unsigned)(ga & 0xFFFFFFFFull);            // global_addr[31:0]
  g0[3] = (unsigned)((ga >> 32) & 0x1FFFFFFull)      // global_addr[56:32]
          | (2u << 30);                              // type = 2 (image)
  i32x8_t g1;
  g1[0] = (int)((2u << 16)                           // data_size = 4B
                | (1u << 20)                         // pad_enable
                | (5u << 22)                         // pad_interval: 64 dwords
                | (1u << 25));                       // pad_amount: 2 dwords
  g1[1] = (int)(64u << 16);                          // tensor_dim0[15:0] @ [31:16]
  g1[2] = (int)(64u << 16);                          // td0 hi=0 | tensor_dim1 lo @ [31:16]
  g1[3] = (int)(64u << 16);                          // td1 hi=0 | tile_dim0=64 @ [31:16]
  g1[4] = (int)(4u | (3u << 16));                    // tile_dim1=4 (h), tile_dim2=3 (d)
  g1[5] = 64;                                        // tensor_dim0_stride lo32 (h stride)
  g1[6] = (int)(4096u << 16);                        // s0 hi=0 | tensor_dim1_stride lo16
  g1[7] = 0;                                         // tensor_dim1_stride hi32 (4096>>16)
  i32x4_t g2;
  g2[0] = 64;                                        // tensor_dim2
  g2[1] = 0;                                         // tensor_dim3 / lds_increment
  g2[2] = 262144;                                    // tensor_dim2_stride lo32 (c stride)
  g2[3] = 0;                                         // s2 hi | tile_dim3=0
  i32x4_t g3 = {0, 0, 0, 0};                         // dims 4 unused
#if defined(__clang_major__) && (__clang_major__ >= 23)
  i32x8_t gx = {0, 0, 0, 0, 0, 0, 0, 0};
  __builtin_amdgcn_tensor_load_to_lds(g0, g1, g2, g3, gx, 0);
#else
  __builtin_amdgcn_tensor_load_to_lds(g0, g1, g2, g3, 0);
#endif
}
#endif  // HAS_TDM

__global__ __launch_bounds__(256) void nca_update_kernel(
    const float* __restrict__ xg, const float* __restrict__ stoch,
    const float* __restrict__ w1, const float* __restrict__ b1,
    const float* __restrict__ w2, const float* __restrict__ b2,
    float* __restrict__ out) {
  // Block = 128 voxels: two w-lines at (b, d0, h0) and (b, d0, h0+1). m = ty*64 + w.
  __shared__ float    xt2[2][3 * 4 * 66];     // double-buffered per-channel halo tile
  __shared__ _Float16 Pbuf[128 * 64];         // perception features, A-matrix layout
  __shared__ _Float16 Hbuf[128 * 128];        // hidden layer (f16), A-matrix layout
  __shared__ float    xc[128 * 16];           // center values per (m, channel)
  __shared__ float    stochs[128];
  __shared__ float    alives[128];

  const int tid  = threadIdx.x;
  const int lane = tid & 31;
  const int wv   = tid >> 5;     // wave id, 0..7
  const int g    = lane >> 4;    // half-wave id

  const int blk = blockIdx.x;
  const int b   = blk >> 11;           // 64*32 = 2048 blocks per batch
  const int d0  = (blk >> 5) & 63;
  const int h0  = (blk & 31) << 1;

  if (tid < 128) {
    const int ty = tid >> 6, w = tid & 63;
    stochs[tid] = stoch[(size_t)b * DD3 + (size_t)d0 * DD2 + (h0 + ty) * DD + w];
  }

  // Zero both halo buffers once: TDM skips the pad slots (halo columns), and
  // out-of-range rows are never written, so zeros implement SAME zero-padding.
  for (int i = tid; i < 2 * 3 * 4 * 66; i += 256) ((float*)xt2)[i] = 0.f;
  __syncthreads();

#if HAS_TDM
  const bool interior = (d0 > 0) && (d0 < 63) && (h0 > 0) && (h0 < 62);
#else
  const bool interior = false;
#endif

  // ---- Perception: one input channel at a time through the halo tile ----
  for (int c = 0; c < C_CH; ++c) {
    float* xt = xt2[c & 1];
    if (interior) {
#if HAS_TDM
      if (wv == 0) {
        // Double-buffered TDM: keep the next channel's DMA in flight.
        if (c == 0) {
          const float* gp = xg + (size_t)(b * C_CH) * DD3 +
                            (size_t)(d0 - 1) * DD2 + (h0 - 1) * DD;
          tdm_load_halo(gp, ((unsigned)(size_t)(void*)&xt2[0][1]));
        }
        if (c + 1 < C_CH) {
          const float* gp = xg + (size_t)(b * C_CH + c + 1) * DD3 +
                            (size_t)(d0 - 1) * DD2 + (h0 - 1) * DD;
          tdm_load_halo(gp, ((unsigned)(size_t)(void*)&xt2[(c + 1) & 1][1]));
          __builtin_amdgcn_s_wait_tensorcnt(1);   // channel c done, c+1 in flight
        } else {
          __builtin_amdgcn_s_wait_tensorcnt(0);
        }
      }
#endif
    } else {
      // Boundary blocks: manual staging with bounds check.
      const float* xch = xg + (size_t)(b * C_CH + c) * DD3;
      for (int i = tid; i < 3 * 4 * 66; i += 256) {
        const int dz = i / 264;             // 4*66
        const int r0 = i - dz * 264;
        const int hy = r0 / 66;
        const int wx = r0 - hy * 66;
        const int gd = d0 + dz - 1;
        const int gh = h0 + hy - 1;
        const int gw = wx - 1;
        float v = 0.f;
        if ((unsigned)gd < 64u && (unsigned)gh < 64u && (unsigned)gw < 64u) {
          const float* ap = xch + (size_t)gd * DD2 + gh * DD + gw;
          v = *ap;
          if (c + 1 < C_CH) __builtin_prefetch(ap + DD3, 0, 1);  // next channel -> L2
        }
        xt[i] = v;
      }
    }
    __syncthreads();   // tile (DMA or manual) visible to all waves

    if (tid < 128) {
      const int ty = tid >> 6, w = tid & 63;
      float a[3][3], bd[3][3];
      float mx = -3.4e38f;
      #pragma unroll
      for (int i = 0; i < 3; ++i) {
        #pragma unroll
        for (int j = 0; j < 3; ++j) {
          const float* q = &xt[(i * 4 + ty + j) * 66 + w];
          const float n0 = q[0], n1 = q[1], n2 = q[2];
          a[i][j]  = n0 + 2.f * n1 + n2;   // w-smooth
          bd[i][j] = n2 - n0;              // w-diff
          if (c == 3) mx = fmaxf(mx, fmaxf(fmaxf(n0, n1), n2));
        }
      }
      const float ident = xt[(4 + ty + 1) * 66 + w + 1];
      const float px = (a[2][0] - a[0][0]) + 2.f * (a[2][1] - a[0][1]) + (a[2][2] - a[0][2]);
      const float py = (a[0][2] - a[0][0]) + 2.f * (a[1][2] - a[1][0]) + (a[2][2] - a[2][0]);
      const float pz = bd[0][0] + 2.f * bd[0][1] + bd[0][2]
                     + 2.f * bd[1][0] + 4.f * bd[1][1] + 2.f * bd[1][2]
                     + bd[2][0] + 2.f * bd[2][1] + bd[2][2];
      _Float16* pp = &Pbuf[tid * 64 + 4 * c];   // 8B-aligned -> ds_store_b64
      pp[0] = (_Float16)ident; pp[1] = (_Float16)px;
      pp[2] = (_Float16)py;    pp[3] = (_Float16)pz;
      xc[tid * 16 + c] = ident;
      if (c == 3) alives[tid] = (mx > 0.1f) ? 1.f : 0.f;
    }
    __syncthreads();   // consumers done before buffer is re-filled
  }

  // ---- Matmul1: H = relu(P[128x64] * W1^T[64x128] + b1). Wave wv -> 16 channels. ----
  const int ch1 = wv * 16 + (lane & 15);
  const float bias1 = b1[ch1];
  const v16h bw0 = load_b_frag_global(w1, wv * 16, 64, 0, lane);
  const v16h bw1 = load_b_frag_global(w1, wv * 16, 64, 32, lane);
  #pragma unroll
  for (int mt = 0; mt < 8; ++mt) {
    const v16h a0 = load_a_frag_lds(Pbuf, mt * 16, 64, 0, lane);
    const v16h a1 = load_a_frag_lds(Pbuf, mt * 16, 64, 32, lane);
    v8f acc = {};
    acc = __builtin_amdgcn_wmma_f32_16x16x32_f16(false, a0, false, bw0, (short)0, acc, false, false);
    acc = __builtin_amdgcn_wmma_f32_16x16x32_f16(false, a1, false, bw1, (short)0, acc, false, false);
    #pragma unroll
    for (int r = 0; r < 8; ++r) {
      const int m = mt * 16 + r + 8 * g;
      const float hv = fmaxf(acc[r] + bias1, 0.f);
      Hbuf[m * 128 + ch1] = (_Float16)hv;
    }
  }
  __syncthreads();

  // ---- Matmul2: dp = H[128x128] * W2^T[128x16] + b2. Wave wv -> M-tile wv. ----
  const int n = lane & 15;
  const float b2v = b2[n];
  v8f acc2 = {};
  #pragma unroll
  for (int kc = 0; kc < 4; ++kc) {
    const v16h af = load_a_frag_lds(Hbuf, wv * 16, 128, kc * 32, lane);
    const v16h bf = load_b_frag_global(w2, 0, 128, kc * 32, lane);
    acc2 = __builtin_amdgcn_wmma_f32_16x16x32_f16(false, af, false, bf, (short)0, acc2, false, false);
  }

  // ---- Fused epilogue: residual update, alive mask / angle mod; transpose via LDS ----
  float* Fbuf = (float*)Pbuf;  // P is dead; reuse as padded [128][17] transpose buffer
  #pragma unroll
  for (int r = 0; r < 8; ++r) {
    const int m = wv * 16 + r + 8 * g;
    const float dpv = acc2[r] + b2v;
    const float maskv = (stochs[m] <= 0.5f) ? 1.f : 0.f;
    const float xn = xc[m * 16 + n] + dpv * maskv;
    float outv;
    if (n == 15) {
      outv = fmodf(xn, TWO_PI_F);
      if (outv < 0.f) outv += TWO_PI_F;
    } else {
      outv = xn * alives[m];
    }
    Fbuf[m * 17 + n] = outv;
  }
  __syncthreads();

  // Coalesced stores: 128 consecutive floats per channel covering both h-lines.
  for (int idx = tid; idx < 16 * 128; idx += 256) {
    const int ch = idx >> 7, m = idx & 127;
    const int ty = m >> 6, w = m & 63;
    out[(size_t)(b * C_CH + ch) * DD3 + (size_t)d0 * DD2 + (h0 + ty) * DD + w] =
        Fbuf[m * 17 + ch];
  }
}

extern "C" void kernel_launch(void* const* d_in, const int* in_sizes, int n_in,
                              void* d_out, int out_size, void* d_ws, size_t ws_size,
                              hipStream_t stream) {
  (void)in_sizes; (void)n_in; (void)out_size; (void)d_ws; (void)ws_size;
  const float* x     = (const float*)d_in[0];
  const float* stoch = (const float*)d_in[1];
  // d_in[2] = percep_kernel: fixed identity/Sobel stack, baked into the kernel.
  const float* w1 = (const float*)d_in[3];
  const float* b1 = (const float*)d_in[4];
  const float* w2 = (const float*)d_in[5];
  const float* b2 = (const float*)d_in[6];
  float* out = (float*)d_out;

  dim3 grid(4 * 64 * 32);   // (b, d, h-pair): 8192 workgroups
  dim3 block(256);          // 8 wave32s
  hipLaunchKernelGGL(nca_update_kernel, grid, block, 0, stream,
                     x, stoch, w1, b1, w2, b2, out);
}
